// RelPartialLearnableDecoderLayer_64914135712024
// MI455X (gfx1250) — matrix operand order repaired and checked by reading, compiled, not run on hardware
//
#include <hip/hip_runtime.h>
#include <hip/hip_bf16.h>

// ---------------------------------------------------------------------------
// Transformer-XL RelPartialLearnableDecoderLayer on MI455X (gfx1250, wave32)
//   - all GEMMs via v_wmma_f32_16x16x32_bf16 (bf16 in, f32 accumulate)
//   - GEMM staging via async Global->LDS (ASYNCcnt) with double buffering
//   - fused flash attention with Transformer-XL relative shift (BD band GEMM)
// ---------------------------------------------------------------------------

typedef __bf16 bf16_t;
typedef __attribute__((ext_vector_type(16))) __bf16 v16bf;
typedef __attribute__((ext_vector_type(8)))  float  v8f;
typedef int v4i_t __attribute__((ext_vector_type(4)));

#define QLEN   2048
#define BSZ    2
#define NHEAD  16
#define DHEAD  64
#define DMODEL 1024
#define DINNER 4096
#define H3     3072           // 3*NHEAD*DHEAD
#define TOK    4096           // QLEN*BSZ
#define NEGMIN (-3.402823466e38f)

#ifndef __has_builtin
#define __has_builtin(x) 0
#endif
#if defined(__AMDGCN__) && \
    __has_builtin(__builtin_amdgcn_global_load_async_to_lds_b128) && \
    __has_builtin(__builtin_amdgcn_s_wait_asynccnt)
#define USE_ASYNC_LDS 1
#endif

#if defined(USE_ASYNC_LDS)
#define ASYNC_WAIT() __builtin_amdgcn_s_wait_asynccnt(0)
#else
#define ASYNC_WAIT()
#endif

union Frag { v16bf v; uint4 q[2]; };

__device__ inline v8f v8f_zero() {
  v8f z = {0.f, 0.f, 0.f, 0.f, 0.f, 0.f, 0.f, 0.f};
  return z;
}

__device__ inline v8f wmma_bf16(Frag a, Frag b, v8f c) {
  return __builtin_amdgcn_wmma_f32_16x16x32_bf16(
      /*neg_a=*/false, a.v, /*neg_b=*/false, b.v,
      /*c_mod=*/(short)0, c, /*reuse_a=*/false, /*reuse_b=*/false);
}

// A fragment (16x32 MxK) from a row-major [M][K] bf16 LDS tile.
// CDNA5 layout: lanes 0-15 row M=lane, K pairs {0..7,16..23}; lanes 16-31 same
// row, K pairs {8..15,24..31}  -> two 16B chunks per lane.
__device__ inline Frag load_a_frag(const bf16_t* base, int row0, int rowStrideB,
                                   int k0, int lane) {
  const int half = (lane >> 4) & 1, l16 = lane & 15;
  const char* r = (const char*)base + (size_t)(row0 + l16) * rowStrideB + k0 * 2;
  Frag f;
  f.q[0] = *(const uint4*)(r + half * 16);
  f.q[1] = *(const uint4*)(r + 32 + half * 16);
  return f;
}

// B fragment (32x16 KxN) from a row-major [N][K] bf16 LDS tile.
// CDNA5 layout: lane holds column N=lane%16, lo lanes K {0..7,16..23}, hi
// lanes K {8..15,24..31} (same order as A fragment).
__device__ inline Frag load_b_frag(const bf16_t* base, int n0, int rowStrideB,
                                   int k0, int lane) {
  const int half = (lane >> 4) & 1, l16 = lane & 15;
  const char* r = (const char*)base + (size_t)(n0 + l16) * rowStrideB + k0 * 2;
  Frag f;
  f.q[0] = *(const uint4*)(r + half * 16);
  f.q[1] = *(const uint4*)(r + 32 + half * 16);
  return f;
}

// ---------------------------------------------------------------------------
// f32 -> bf16 pack
// ---------------------------------------------------------------------------
__global__ void pack_bf16(const float* __restrict__ x, bf16_t* __restrict__ y, int n) {
  for (int i = blockIdx.x * blockDim.x + threadIdx.x; i < n;
       i += gridDim.x * blockDim.x)
    y[i] = (bf16_t)x[i];
}

// ---------------------------------------------------------------------------
// C[M,N] f32 = A[M,K] bf16 * B[N,K] bf16 ^T   (M%128==0, N%128==0, K%32==0)
// block tile 128x128, 8 waves (4x2), wave tile 32x64, K-step 32.
// Double-buffered LDS; tile k+1 streamed with async Global->LDS while tile k
// is consumed by WMMA. One barrier per K-step (compiler emits s_wait_dscnt
// before s_barrier_signal, covering the LDS read/write handoff).
// ---------------------------------------------------------------------------
__global__ __launch_bounds__(256)
void gemm_bf16_nt(const bf16_t* __restrict__ A, const bf16_t* __restrict__ B,
                  float* __restrict__ C, int M, int N, int K) {
  __shared__ __align__(16) bf16_t sA[2][128 * 32];
  __shared__ __align__(16) bf16_t sB[2][128 * 32];
  const int tid = threadIdx.x, lane = tid & 31, wave = tid >> 5;
  const int wm = wave & 3, wn = wave >> 2;
  const int m0 = blockIdx.y * 128, n0 = blockIdx.x * 128;
  const int l16 = lane & 15, half = (lane >> 4) & 1;
  (void)M;

  v8f acc[2][4];
  for (int mi = 0; mi < 2; ++mi)
    for (int ni = 0; ni < 4; ++ni) acc[mi][ni] = v8f_zero();

  // 512 16-byte chunks per 128x32 tile -> 2 chunks per thread per matrix
  auto issue = [&](int k0, int buf) {
    for (int u = 0; u < 2; ++u) {
      int chunk = tid + 256 * u;
      int row = chunk >> 2, c4 = (chunk & 3) * 8;
      const bf16_t* ga = A + (size_t)(m0 + row) * K + k0 + c4;
      const bf16_t* gb = B + (size_t)(n0 + row) * K + k0 + c4;
      bf16_t* la = &sA[buf][row * 32 + c4];
      bf16_t* lb = &sB[buf][row * 32 + c4];
#if defined(USE_ASYNC_LDS)
      __builtin_amdgcn_global_load_async_to_lds_b128(
          (v4i_t*)ga, (v4i_t*)la, 0, 0);
      __builtin_amdgcn_global_load_async_to_lds_b128(
          (v4i_t*)gb, (v4i_t*)lb, 0, 0);
#else
      *(uint4*)la = *(const uint4*)ga;
      *(uint4*)lb = *(const uint4*)gb;
      if (k0 + 32 < K) {   // lowers to global_prefetch_b8 on gfx1250
        __builtin_prefetch(ga + 32, 0, 3);
        __builtin_prefetch(gb + 32, 0, 3);
      }
#endif
    }
  };

  issue(0, 0);
  const int nk = K >> 5;
  for (int ki = 0; ki < nk; ++ki) {
    const int buf = ki & 1;
    ASYNC_WAIT();
    __syncthreads();
    if (ki + 1 < nk) issue((ki + 1) << 5, buf ^ 1);

    Frag af[2], bfr[4];
    for (int mi = 0; mi < 2; ++mi)
      af[mi] = load_a_frag(sA[buf], wm * 32 + mi * 16, 64, 0, lane);
    for (int ni = 0; ni < 4; ++ni)
      bfr[ni] = load_b_frag(sB[buf], wn * 64 + ni * 16, 64, 0, lane);
    for (int mi = 0; mi < 2; ++mi)
      for (int ni = 0; ni < 4; ++ni)
        acc[mi][ni] = wmma_bf16(af[mi], bfr[ni], acc[mi][ni]);
  }

  for (int mi = 0; mi < 2; ++mi)
    for (int ni = 0; ni < 4; ++ni)
      for (int p = 0; p < 8; ++p) {
        int r = m0 + wm * 32 + mi * 16 + p + 8 * half;
        int c = n0 + wn * 64 + ni * 16 + l16;
        C[(size_t)r * N + c] = acc[mi][ni][p];
      }
}

// ---------------------------------------------------------------------------
// Fused flash attention with Transformer-XL relative shift.
// grid = (QLEN/64, BSZ, NHEAD), block = 128 (4 waves); each wave owns 16
// query rows. BD[i,j] = Qrr[i] . RK[q-1-i+j]; per 64x64 tile that is a
// 128-row RK band, evaluated with WMMA into per-wave LDS and gathered
// diagonally. Causal mask + online softmax + PV accumulation kept on-chip.
// ---------------------------------------------------------------------------
__global__ __launch_bounds__(128)
void attn_flash(const float* __restrict__ Wheads, const float* __restrict__ RK,
                const float* __restrict__ rwb, const float* __restrict__ rrb,
                bf16_t* __restrict__ attnV) {
  const int i0 = blockIdx.x * 64;
  const int b  = blockIdx.y;
  const int n  = blockIdx.z;
  const int tid = threadIdx.x, lane = tid & 31, wave = tid >> 5;
  const int l16 = lane & 15, half = (lane >> 4) & 1;
  const float scale = 0.125f;   // 1/sqrt(DHEAD), folded into Q staging

  __shared__ __align__(16) char smem[65536];
  bf16_t* sK  = (bf16_t*)(smem);                         // [64][64] (also Q staging)
  bf16_t* sVt = (bf16_t*)(smem + 8192);                  // [64][64] V transposed
  bf16_t* sRK = (bf16_t*)(smem + 16384);                 // [128][64] RK band
  float*  sGw = (float*)(smem + 32768 + wave * 8192);    // per-wave [16][128] G
  bf16_t* sPw = (bf16_t*)(smem + 32768 + wave * 8192);   // per-wave [16][64] P (aliases dead G)

  // ---- stage Q(+rw bias) and Q(+rr bias), prescaled, keep frags in regs ----
  Frag qrw[2], qrr[2];
  for (int pass = 0; pass < 2; ++pass) {
    const float* bias = (pass == 0) ? rwb : rrb;
    for (int e = tid; e < 4096; e += 128) {
      int a = e >> 6, d = e & 63;
      int t = (i0 + a) * BSZ + b;
      float v = Wheads[(size_t)t * H3 + n * DHEAD + d] + bias[n * DHEAD + d];
      sK[a * 64 + d] = (bf16_t)(v * scale);
    }
    __syncthreads();
    Frag* dst = (pass == 0) ? qrw : qrr;
    dst[0] = load_a_frag(sK, wave * 16, 128, 0, lane);
    dst[1] = load_a_frag(sK, wave * 16, 128, 32, lane);
    __syncthreads();
  }

  v8f accO[4];
  float m[8], l[8];
  for (int di = 0; di < 4; ++di) accO[di] = v8f_zero();
  for (int p = 0; p < 8; ++p) { m[p] = NEGMIN; l[p] = 0.f; }

  for (int j0 = 0; j0 <= i0; j0 += 64) {
    // ---- stage K tile, V^T tile, RK band ----
    for (int e = tid; e < 4096; e += 128) {
      int jr = e >> 6, d = e & 63;
      int t = (j0 + jr) * BSZ + b;
      sK[jr * 64 + d]  = (bf16_t)Wheads[(size_t)t * H3 + DMODEL + n * DHEAD + d];
      sVt[d * 64 + jr] = (bf16_t)Wheads[(size_t)t * H3 + 2 * DMODEL + n * DHEAD + d];
    }
    const int pb = j0 - i0 + QLEN - 64;  // band base: p = q-1-(i0+63)+j0
    for (int e = tid; e < 8192; e += 128) {
      int pr = e >> 6, d = e & 63;
      int pabs = pb + pr;
      float v = (pabs < QLEN) ? RK[(size_t)pabs * DMODEL + n * DHEAD + d] : 0.f;
      sRK[pr * 64 + d] = (bf16_t)v;
    }
    __syncthreads();

    // ---- G = Qrr x RKband^T (only the 5 band-column tiles this wave reads) ----
    for (int ni = 3 - wave; ni <= 7 - wave; ++ni) {
      v8f g = v8f_zero();
      g = wmma_bf16(qrr[0], load_b_frag(sRK, ni * 16, 128, 0, lane), g);
      g = wmma_bf16(qrr[1], load_b_frag(sRK, ni * 16, 128, 32, lane), g);
      for (int p = 0; p < 8; ++p)
        sGw[(p + 8 * half) * 128 + ni * 16 + l16] = g[p];
    }

    // ---- AC = Qrw x K^T ----
    v8f accS[4];
    for (int ni = 0; ni < 4; ++ni) {
      v8f s = v8f_zero();
      s = wmma_bf16(qrw[0], load_b_frag(sK, ni * 16, 128, 0, lane), s);
      s = wmma_bf16(qrw[1], load_b_frag(sK, ni * 16, 128, 32, lane), s);
      accS[ni] = s;
    }

    // ---- S = AC + BD(gathered diagonal), causal mask ----
    float sv[4][8];
    for (int ni = 0; ni < 4; ++ni)
      for (int p = 0; p < 8; ++p) {
        int aloc = p + 8 * half;
        int ablk = wave * 16 + aloc;
        int c = ni * 16 + l16;
        float s = accS[ni][p] + sGw[aloc * 128 + (63 + c - ablk)];
        sv[ni][p] = ((j0 + c) > (i0 + ablk)) ? NEGMIN : s;
      }

    // ---- online softmax stats (row reductions across 16-lane halves) ----
    float fs[8], mn[8];
    for (int p = 0; p < 8; ++p) {
      float v = fmaxf(fmaxf(sv[0][p], sv[1][p]), fmaxf(sv[2][p], sv[3][p]));
      for (int msk = 1; msk < 16; msk <<= 1) v = fmaxf(v, __shfl_xor(v, msk, 32));
      mn[p] = fmaxf(m[p], v);
      fs[p] = __expf(m[p] - mn[p]);
      m[p]  = mn[p];
    }

    __syncthreads();  // all G reads done before P aliases the slab

    for (int p = 0; p < 8; ++p) {
      float rs = 0.f;
      for (int ni = 0; ni < 4; ++ni) {
        float pv = __expf(sv[ni][p] - mn[p]);
        sPw[(p + 8 * half) * 64 + ni * 16 + l16] = (bf16_t)pv;
        rs += pv;
      }
      for (int msk = 1; msk < 16; msk <<= 1) rs += __shfl_xor(rs, msk, 32);
      l[p] = l[p] * fs[p] + rs;
      for (int di = 0; di < 4; ++di) accO[di][p] *= fs[p];
    }

    // ---- O += P x V ----
    Frag pf0 = load_a_frag(sPw, 0, 128, 0, lane);
    Frag pf1 = load_a_frag(sPw, 0, 128, 32, lane);
    for (int di = 0; di < 4; ++di) {
      accO[di] = wmma_bf16(pf0, load_b_frag(sVt, di * 16, 128, 0, lane), accO[di]);
      accO[di] = wmma_bf16(pf1, load_b_frag(sVt, di * 16, 128, 32, lane), accO[di]);
    }
    __syncthreads();
  }

  // ---- normalize + store attn_vec (bf16, feeds out-proj GEMM) ----
  for (int di = 0; di < 4; ++di)
    for (int p = 0; p < 8; ++p) {
      int ablk = wave * 16 + p + 8 * half;
      int t = (i0 + ablk) * BSZ + b;
      int d = di * 16 + l16;
      attnV[(size_t)t * DMODEL + n * DHEAD + d] = (bf16_t)(accO[di][p] / l[p]);
    }
}

// ---------------------------------------------------------------------------
// out = LayerNorm(resid + y [+ bias]) * g + beta ; optional f32 / bf16 outs
// one row (DMODEL=1024) per block, 256 threads x 4 cols
// ---------------------------------------------------------------------------
__global__ __launch_bounds__(256)
void ln_fused(const float* __restrict__ resid, const float* __restrict__ y,
              const float* __restrict__ bias,
              const float* __restrict__ g, const float* __restrict__ beta,
              float* __restrict__ out_f, bf16_t* __restrict__ out_bf) {
  const int row = blockIdx.x, tid = threadIdx.x;
  __shared__ float red[256];
  float x[4];
  for (int u = 0; u < 4; ++u) {
    int c = tid + u * 256;
    float v = resid[(size_t)row * DMODEL + c] + y[(size_t)row * DMODEL + c];
    if (bias) v += bias[c];
    x[u] = v;
  }
  float s1 = x[0] + x[1] + x[2] + x[3];
  red[tid] = s1; __syncthreads();
  for (int o = 128; o > 0; o >>= 1) { if (tid < o) red[tid] += red[tid + o]; __syncthreads(); }
  float mean = red[0] * (1.f / DMODEL);
  __syncthreads();
  float s2 = x[0]*x[0] + x[1]*x[1] + x[2]*x[2] + x[3]*x[3];
  red[tid] = s2; __syncthreads();
  for (int o = 128; o > 0; o >>= 1) { if (tid < o) red[tid] += red[tid + o]; __syncthreads(); }
  float var = red[0] * (1.f / DMODEL) - mean * mean;
  float rsig = rsqrtf(var + 1e-5f);
  for (int u = 0; u < 4; ++u) {
    int c = tid + u * 256;
    float ov = (x[u] - mean) * rsig * g[c] + beta[c];
    if (out_f)  out_f[(size_t)row * DMODEL + c] = ov;
    if (out_bf) out_bf[(size_t)row * DMODEL + c] = (bf16_t)ov;
  }
}

// ---------------------------------------------------------------------------
// hidden = relu(h + b1) packed to bf16  (cols = DINNER, power of two)
// ---------------------------------------------------------------------------
__global__ void bias_relu_bf16(const float* __restrict__ h, const float* __restrict__ b,
                               bf16_t* __restrict__ o, int n) {
  for (int i = blockIdx.x * blockDim.x + threadIdx.x; i < n;
       i += gridDim.x * blockDim.x) {
    float v = h[i] + b[i & (DINNER - 1)];
    o[i] = (bf16_t)fmaxf(v, 0.f);
  }
}

// ---------------------------------------------------------------------------
extern "C" void kernel_launch(void* const* d_in, const int* in_sizes, int n_in,
                              void* d_out, int out_size, void* d_ws, size_t ws_size,
                              hipStream_t stream) {
  (void)in_sizes; (void)n_in; (void)out_size; (void)ws_size;
  const float* dec_inp   = (const float*)d_in[0];
  const float* r         = (const float*)d_in[1];
  const float* rwb       = (const float*)d_in[2];
  const float* rrb       = (const float*)d_in[3];
  const float* qkv_w     = (const float*)d_in[4];
  const float* rnet_w    = (const float*)d_in[5];
  const float* o_w       = (const float*)d_in[6];
  const float* ln_attn_g = (const float*)d_in[7];
  const float* ln_attn_b = (const float*)d_in[8];
  const float* w1        = (const float*)d_in[9];
  const float* b1        = (const float*)d_in[10];
  const float* w2        = (const float*)d_in[11];
  const float* b2        = (const float*)d_in[12];
  const float* ln_ff_g   = (const float*)d_in[13];
  const float* ln_ff_b   = (const float*)d_in[14];
  // d_in[15] = attn_mask (bool) : causal mask computed analytically in-kernel

  char* base = (char*)d_ws;
  size_t off = 0;
  auto alloc = [&](size_t bytes) -> void* {
    void* p = base + off;
    off += (bytes + 255) & ~(size_t)255;
    return p;
  };
  bf16_t* Xbf     = (bf16_t*)alloc((size_t)TOK * DMODEL * 2);
  bf16_t* qkvWbf  = (bf16_t*)alloc((size_t)H3 * DMODEL * 2);
  bf16_t* rbf     = (bf16_t*)alloc((size_t)QLEN * DMODEL * 2);
  bf16_t* rnetWbf = (bf16_t*)alloc((size_t)DMODEL * DMODEL * 2);
  bf16_t* oWbf    = (bf16_t*)alloc((size_t)DMODEL * DMODEL * 2);
  bf16_t* w1bf    = (bf16_t*)alloc((size_t)DINNER * DMODEL * 2);
  bf16_t* w2bf    = (bf16_t*)alloc((size_t)DMODEL * DINNER * 2);
  float*  Wheads  = (float*) alloc((size_t)TOK * H3 * 4);
  float*  RK      = (float*) alloc((size_t)QLEN * DMODEL * 4);
  bf16_t* attnV   = (bf16_t*)alloc((size_t)TOK * DMODEL * 2);
  float*  attnOut = (float*) alloc((size_t)TOK * DMODEL * 4);
  float*  out1f   = (float*) alloc((size_t)TOK * DMODEL * 4);
  bf16_t* out1bf  = (bf16_t*)alloc((size_t)TOK * DMODEL * 2);
  float*  hidf    = (float*) alloc((size_t)TOK * DINNER * 4);
  bf16_t* hidbf   = (bf16_t*)alloc((size_t)TOK * DINNER * 2);
  float*  coref   = (float*) alloc((size_t)TOK * DMODEL * 4);

  // ---- pack inputs/weights to bf16 ----
  pack_bf16<<<1024, 256, 0, stream>>>(dec_inp, Xbf, TOK * DMODEL);
  pack_bf16<<<1024, 256, 0, stream>>>(qkv_w, qkvWbf, H3 * DMODEL);
  pack_bf16<<<1024, 256, 0, stream>>>(r, rbf, QLEN * DMODEL);
  pack_bf16<<<1024, 256, 0, stream>>>(rnet_w, rnetWbf, DMODEL * DMODEL);
  pack_bf16<<<1024, 256, 0, stream>>>(o_w, oWbf, DMODEL * DMODEL);
  pack_bf16<<<1024, 256, 0, stream>>>(w1, w1bf, DINNER * DMODEL);
  pack_bf16<<<1024, 256, 0, stream>>>(w2, w2bf, DMODEL * DINNER);

  // ---- QKV and relative-key projections ----
  gemm_bf16_nt<<<dim3(H3 / 128, TOK / 128), 256, 0, stream>>>(
      Xbf, qkvWbf, Wheads, TOK, H3, DMODEL);
  gemm_bf16_nt<<<dim3(DMODEL / 128, QLEN / 128), 256, 0, stream>>>(
      rbf, rnetWbf, RK, QLEN, DMODEL, DMODEL);

  // ---- fused attention (AC + shifted BD + mask + softmax + PV) ----
  attn_flash<<<dim3(QLEN / 64, BSZ, NHEAD), 128, 0, stream>>>(
      Wheads, RK, rwb, rrb, attnV);

  // ---- output projection, residual + LN ----
  gemm_bf16_nt<<<dim3(DMODEL / 128, TOK / 128), 256, 0, stream>>>(
      attnV, oWbf, attnOut, TOK, DMODEL, DMODEL);
  ln_fused<<<TOK, 256, 0, stream>>>(dec_inp, attnOut, nullptr,
                                    ln_attn_g, ln_attn_b, out1f, out1bf);

  // ---- positionwise FFN, residual + LN ----
  gemm_bf16_nt<<<dim3(DINNER / 128, TOK / 128), 256, 0, stream>>>(
      out1bf, w1bf, hidf, TOK, DINNER, DMODEL);
  bias_relu_bf16<<<2048, 256, 0, stream>>>(hidf, b1, hidbf, TOK * DINNER);
  gemm_bf16_nt<<<dim3(DMODEL / 128, TOK / 128), 256, 0, stream>>>(
      hidbf, w2bf, coref, TOK, DMODEL, DINNER);
  ln_fused<<<TOK, 256, 0, stream>>>(out1f, coref, b2,
                                    ln_ff_g, ln_ff_b, (float*)d_out, nullptr);
}
